// LiDARBEV_36764920054282
// MI455X (gfx1250) — matrix operand order repaired and checked by reading, not compiled, not run on hardware
//
#include <hip/hip_runtime.h>

typedef __attribute__((ext_vector_type(16))) _Float16 v16h;
typedef __attribute__((ext_vector_type(8)))  _Float16 v8h;
typedef __attribute__((ext_vector_type(8)))  float    v8f;

#define DD 128
#define HH 128
#define WW 32
#define NB 2
#define NPTS 100000
#define PP (NB*DD*HH*WW)          /* 1,048,576 spatial positions          */
#define INV_N (1.0f/1048576.0f)
#define EPS_BN 1e-5f

/* ---------------- workspace layout (bytes) ----------------
   0        : minb (6 f32)
   256      : stats (3 layers x [64 sum | 64 sumsq] f32 = 384 f32)
   4096     : wp2  repacked conv2 weights f16 (55,296 halves)
   135168   : wp3  repacked conv3 weights f16 (110,592 halves)
   397312   : grid f32 (PP floats, 4 MB)
   4591616  : buf0 f16 (PP*64 halves, 134 MB)
   138809344: buf1 f16 (PP*64 halves, 134 MB)                  */
#define OFF_STATS 256
#define OFF_WP2   4096
#define OFF_WP3   135168
#define OFF_GRID  397312
#define OFF_BUF0  4591616
#define OFF_BUF1  138809344

// ---------------------------------------------------------------- zero
__global__ __launch_bounds__(256) void k_zero(float* grid, float* stats) {
  int i = blockIdx.x * 256 + threadIdx.x;
  if (i < PP) grid[i] = 0.f;
  if (i < 384) stats[i] = 0.f;
}

// ------------------------------------------------- per-(b,coord) min reduce
__global__ __launch_bounds__(256) void k_minred(const float* __restrict__ pc,
                                                float* __restrict__ minb) {
  __shared__ float sm[256];
  int bc = blockIdx.x;                       // 0..5 = b*3+coord
  const float* p = pc + (long)bc * NPTS;
  float m = 3.402823466e38f;
  for (int i = threadIdx.x; i < NPTS; i += 256) {
    float v = p[i];
    v = (v != v) ? 0.f : v;                  // NaN -> 0 (matches reference)
    m = fminf(m, v);
  }
  sm[threadIdx.x] = m;
  __syncthreads();
  for (int s = 128; s > 0; s >>= 1) {
    if (threadIdx.x < s) sm[threadIdx.x] = fminf(sm[threadIdx.x], sm[threadIdx.x + s]);
    __syncthreads();
  }
  if (threadIdx.x == 0) minb[bc] = sm[0];
}

// ---------------------------------------------------------------- scatter
__global__ __launch_bounds__(256) void k_scatter(const float* __restrict__ pc,
                                                 const float* __restrict__ minb,
                                                 float* __restrict__ grid) {
  int tid = blockIdx.x * 256 + threadIdx.x;
  if (tid >= NB * NPTS) return;
  int b = tid / NPTS;
  int n = tid - b * NPTS;
  float px = pc[((long)b * 3 + 0) * NPTS + n]; px = (px != px) ? 0.f : px;
  float py = pc[((long)b * 3 + 1) * NPTS + n]; py = (py != py) ? 0.f : py;
  float pz = pc[((long)b * 3 + 2) * NPTS + n]; pz = (pz != pz) ? 0.f : pz;
  int ix = (int)floorf((px - minb[b * 3 + 0]) * 16.0f);   // 1/0.0625
  int iy = (int)floorf((py - minb[b * 3 + 1]) * 16.0f);
  int iz = (int)floorf((pz - minb[b * 3 + 2]) * 16.0f);
  ix = min(max(ix, 0), DD - 1);
  iy = min(max(iy, 0), HH - 1);
  iz = min(max(iz, 0), WW - 1);
  grid[((b * DD + ix) * HH + iy) * WW + iz] = 1.0f;       // all writers store 1.0
}

// ------------------------------------------- conv1 (1->32) scalar, out NDHWC f16
__global__ __launch_bounds__(256) void k_conv1(const float* __restrict__ grid,
                                               const float* __restrict__ w1,
                                               const float* __restrict__ b1,
                                               _Float16* __restrict__ x1) {
  __shared__ float sw[32 * 27];
  __shared__ float sb[32];
  for (int i = threadIdx.x; i < 32 * 27; i += 256) sw[i] = w1[i];
  if (threadIdx.x < 32) sb[threadIdx.x] = b1[threadIdx.x];
  __syncthreads();
  int pos = blockIdx.x * 256 + threadIdx.x;               // < PP exactly
  int w = pos & 31, h = (pos >> 5) & 127, d = (pos >> 12) & 127, b = pos >> 19;
  float gv[27];
#pragma unroll
  for (int t = 0; t < 27; ++t) {
    int kd = t / 9, kh = (t / 3) % 3, kw = t % 3;
    int di = d + kd - 1, hi = h + kh - 1, wi = w + kw - 1;
    bool ok = ((unsigned)di < 128u) & ((unsigned)hi < 128u) & ((unsigned)wi < 32u);
    gv[t] = ok ? grid[((b * DD + di) * HH + hi) * WW + wi] : 0.f;
  }
  long ob = (long)pos * 32;
  for (int c = 0; c < 32; c += 8) {
    v8h r;
#pragma unroll
    for (int q = 0; q < 8; ++q) {
      float acc = sb[c + q];
#pragma unroll
      for (int t = 0; t < 27; ++t) acc += gv[t] * sw[(c + q) * 27 + t];
      r[q] = (_Float16)acc;
    }
    *(v8h*)(x1 + ob + c) = r;
  }
}

// ----------------------------------------------- per-channel sum / sumsq stats
__global__ __launch_bounds__(256) void k_stats(const _Float16* __restrict__ y,
                                               float* __restrict__ sum,
                                               float* __restrict__ sumsq,
                                               int C, long total) {
  __shared__ float s1[256], s2[256];
  int tid = threadIdx.x;
  long stride = (long)gridDim.x * 256;
  float a = 0.f, b = 0.f;
  for (long i = (long)blockIdx.x * 256 + tid; i < total; i += stride) {
    float v = (float)y[i];                   // C | 256 and C | stride, so
    a += v; b += v * v;                      // this thread's channel == tid % C
  }
  s1[tid] = a; s2[tid] = b;
  __syncthreads();
  if (tid < C) {
    float sa = 0.f, sb = 0.f;
    for (int i = tid; i < 256; i += C) { sa += s1[i]; sb += s2[i]; }
    atomicAdd(&sum[tid], sa);
    atomicAdd(&sumsq[tid], sb);
  }
}

// --------------------------------------------------- bn + relu, in place f16
__global__ __launch_bounds__(256) void k_bnrelu(_Float16* __restrict__ y,
                                                const float* __restrict__ sum,
                                                const float* __restrict__ sumsq,
                                                const float* __restrict__ g,
                                                const float* __restrict__ be,
                                                int cmask, long total) {
  long stride = (long)gridDim.x * 256;
  for (long i = (long)blockIdx.x * 256 + threadIdx.x; i < total; i += stride) {
    int c = (int)(i & (long)cmask);
    float mean = sum[c] * INV_N;
    float var = sumsq[c] * INV_N - mean * mean;
    float sc = g[c] * rsqrtf(var + EPS_BN);
    float v = (float)y[i] * sc + (be[c] - mean * sc);
    y[i] = (_Float16)fmaxf(v, 0.f);
  }
}

// -------------------------- repack weights into exact WMMA A-fragment layout
__global__ __launch_bounds__(256) void k_repack(const float* __restrict__ w,
                                                _Float16* __restrict__ wp,
                                                int Cin, int nTot) {
  int i = blockIdx.x * 256 + threadIdx.x;
  if (i >= nTot) return;
  int j    = i & 15;            // half index within lane's v16h
  int lane = (i >> 4) & 31;
  int nCb  = Cin >> 5;
  int blk  = i >> 9;            // 512-half A-fragment block
  int cb   = blk % nCb;
  int rest = blk / nCb;
  int t    = rest % 27;
  int mt   = rest / 27;
  int m    = mt * 16 + (lane & 15);
  // contiguous K layout (matches B fill): lane<16 -> K 0..15; lane>=16 -> K 16..31
  int k = ((lane >> 4) << 4) + j;
  int c = cb * 32 + k;
  wp[i] = (_Float16)w[(m * Cin + c) * 27 + t];
}

// --------------------- implicit-GEMM 3x3x3 conv via v_wmma_f32_16x16x32_f16
// x: NDHWC f16 (C = CIN), y: NDHWC f16 (C = 64).
// One wave = one full W line (2 x 16 positions) x ALL 64 out-channels:
//   8 independent accumulators; per (tap, cb) step: 2 B-loads + 4 A-loads
//   feed 8 WMMAs -> 0.75 KB of wave loads per WMMA and 8-way WMMA ILP.
template <int CIN>
__global__ __launch_bounds__(256) void k_conv_wmma(const _Float16* __restrict__ x,
                                                   const _Float16* __restrict__ wp,
                                                   const float* __restrict__ bias,
                                                   _Float16* __restrict__ y) {
  constexpr int NCB = CIN >> 5;
  constexpr int ASTEP = 27 * NCB * 32;               // mt stride in v16h units
  int lane = threadIdx.x & 31;
  int s = (blockIdx.x << 3) + (threadIdx.x >> 5);    // W-line id, < 32768 exact
  int h = s & 127;
  int d = (s >> 7) & 127;
  int b = s >> 14;
  int col   = lane & 15;
  int hi16  = lane >> 4;
  int choff = hi16 << 4;                             // B-frag: lane>=16 holds K 16..31
  const v16h* __restrict__ wv16 = (const v16h*)wp;
  v8f a00 = {}, a01 = {}, a10 = {}, a11 = {};        // acc[mt][tile]
  v8f a20 = {}, a21 = {}, a30 = {}, a31 = {};
#pragma unroll
  for (int t = 0; t < 27; ++t) {
    const int kd = t / 9, kh = (t / 3) % 3, kw = t % 3;
    int di = d + kd - 1, hi = h + kh - 1;
    bool rowok = ((unsigned)di < 128u) & ((unsigned)hi < 128u);
    int wi0 = col + kw - 1;                          // tile0 input W (-1..16)
    int wi1 = wi0 + 16;                              // tile1 input W (15..32)
    bool v0 = rowok & (wi0 >= 0);
    bool v1 = rowok & (wi1 < WW);
    const _Float16* xrow =
        x + ((long)(b * DD + di) * HH + hi) * (long)(WW * CIN) + choff;
#pragma unroll
    for (int cb = 0; cb < NCB; ++cb) {
      v16h bf0 = {}, bf1 = {};
      if (v0) bf0 = *(const v16h*)(xrow + wi0 * CIN + cb * 32);   // 32B/lane
      if (v1) bf1 = *(const v16h*)(xrow + wi1 * CIN + cb * 32);
      const v16h* ap = wv16 + (t * NCB + cb) * 32 + lane;
      v16h A0 = ap[0 * ASTEP];
      v16h A1 = ap[1 * ASTEP];
      v16h A2 = ap[2 * ASTEP];
      v16h A3 = ap[3 * ASTEP];
      a00 = __builtin_amdgcn_wmma_f32_16x16x32_f16(false, A0, false, bf0, (short)0, a00, false, false);
      a01 = __builtin_amdgcn_wmma_f32_16x16x32_f16(false, A0, false, bf1, (short)0, a01, false, false);
      a10 = __builtin_amdgcn_wmma_f32_16x16x32_f16(false, A1, false, bf0, (short)0, a10, false, false);
      a11 = __builtin_amdgcn_wmma_f32_16x16x32_f16(false, A1, false, bf1, (short)0, a11, false, false);
      a20 = __builtin_amdgcn_wmma_f32_16x16x32_f16(false, A2, false, bf0, (short)0, a20, false, false);
      a21 = __builtin_amdgcn_wmma_f32_16x16x32_f16(false, A2, false, bf1, (short)0, a21, false, false);
      a30 = __builtin_amdgcn_wmma_f32_16x16x32_f16(false, A3, false, bf0, (short)0, a30, false, false);
      a31 = __builtin_amdgcn_wmma_f32_16x16x32_f16(false, A3, false, bf1, (short)0, a31, false, false);
    }
  }
  // D layout: VGPR i = M row (i + 8*hi16), column = lane&15 -> 8 consecutive channels
  long pbase = ((long)(b * DD + d) * HH + h) * (long)(WW * 64);
  int ocbase = hi16 << 3;
  v8f accs[4][2] = {{a00, a01}, {a10, a11}, {a20, a21}, {a30, a31}};
#pragma unroll
  for (int mt = 0; mt < 4; ++mt) {
    int oc = mt * 16 + ocbase;
#pragma unroll
    for (int u = 0; u < 2; ++u) {
      v8h r;
#pragma unroll
      for (int i = 0; i < 8; ++i) r[i] = (_Float16)(accs[mt][u][i] + bias[oc + i]);
      *(v8h*)(y + pbase + (long)(u * 16 + col) * 64 + oc) = r;
    }
  }
}

// ------------------- final: bn + max over D + relu -> (B,64,H,W) f32 NCHW
__global__ __launch_bounds__(256) void k_finalmax(const _Float16* __restrict__ y,
                                                  const float* __restrict__ sum,
                                                  const float* __restrict__ sumsq,
                                                  const float* __restrict__ g,
                                                  const float* __restrict__ be,
                                                  float* __restrict__ out) {
  int tid = blockIdx.x * 256 + threadIdx.x;          // < NB*HH*WW*64 exactly
  int o = tid & 63;
  int w = (tid >> 6) & 31;
  int h = (tid >> 11) & 127;
  int b = tid >> 18;
  float mean = sum[o] * INV_N;
  float var  = sumsq[o] * INV_N - mean * mean;
  float sc = g[o] * rsqrtf(var + EPS_BN);
  float sh = be[o] - mean * sc;
  float m = -3.402823466e38f;
  for (int d = 0; d < DD; ++d) {
    float v = (float)y[(((long)(b * DD + d) * HH + h) * WW + w) * 64 + o] * sc + sh;
    m = fmaxf(m, v);
  }
  out[((b * 64 + o) * HH + h) * WW + w] = fmaxf(m, 0.f);   // relu commutes with max
}

extern "C" void kernel_launch(void* const* d_in, const int* in_sizes, int n_in,
                              void* d_out, int out_size, void* d_ws, size_t ws_size,
                              hipStream_t stream) {
  const float* pc  = (const float*)d_in[0];
  const float* w1  = (const float*)d_in[1];
  const float* b1  = (const float*)d_in[2];
  const float* g1  = (const float*)d_in[3];
  const float* be1 = (const float*)d_in[4];
  const float* w2  = (const float*)d_in[5];
  const float* b2  = (const float*)d_in[6];
  const float* g2  = (const float*)d_in[7];
  const float* be2 = (const float*)d_in[8];
  const float* w3  = (const float*)d_in[9];
  const float* b3  = (const float*)d_in[10];
  const float* g3  = (const float*)d_in[11];
  const float* be3 = (const float*)d_in[12];
  float* outp = (float*)d_out;

  char* ws = (char*)d_ws;
  float*    minb  = (float*)ws;
  float*    stats = (float*)(ws + OFF_STATS);
  _Float16* wp2   = (_Float16*)(ws + OFF_WP2);
  _Float16* wp3   = (_Float16*)(ws + OFF_WP3);
  float*    grid  = (float*)(ws + OFF_GRID);
  _Float16* buf0  = (_Float16*)(ws + OFF_BUF0);
  _Float16* buf1  = (_Float16*)(ws + OFF_BUF1);

  const long n32 = (long)PP * 32, n64 = (long)PP * 64;

  k_zero<<<PP / 256, 256, 0, stream>>>(grid, stats);
  k_minred<<<6, 256, 0, stream>>>(pc, minb);
  k_scatter<<<(NB * NPTS + 255) / 256, 256, 0, stream>>>(pc, minb, grid);

  k_conv1<<<PP / 256, 256, 0, stream>>>(grid, w1, b1, buf0);
  k_stats<<<2048, 256, 0, stream>>>(buf0, stats + 0, stats + 64, 32, n32);
  k_bnrelu<<<2048, 256, 0, stream>>>(buf0, stats + 0, stats + 64, g1, be1, 31, n32);

  k_repack<<<(55296 + 255) / 256, 256, 0, stream>>>(w2, wp2, 32, 55296);
  k_conv_wmma<32><<<4096, 256, 0, stream>>>(buf0, wp2, b2, buf1);
  k_stats<<<4096, 256, 0, stream>>>(buf1, stats + 128, stats + 192, 64, n64);
  k_bnrelu<<<4096, 256, 0, stream>>>(buf1, stats + 128, stats + 192, g2, be2, 63, n64);

  k_repack<<<(110592 + 255) / 256, 256, 0, stream>>>(w3, wp3, 64, 110592);
  k_conv_wmma<64><<<4096, 256, 0, stream>>>(buf1, wp3, b3, buf0);
  k_stats<<<4096, 256, 0, stream>>>(buf0, stats + 256, stats + 320, 64, n64);

  k_finalmax<<<(NB * HH * WW * 64) / 256, 256, 0, stream>>>(
      buf0, stats + 256, stats + 320, g3, be3, outp);
}
